// SpinalCordCircuit_230630_unified_15006615734205
// MI455X (gfx1250) — matrix operand order repaired and checked by reading, not compiled, not run on hardware
//
#include <hip/hip_runtime.h>
#include <hip/hip_fp16.h>

// ---------------------------------------------------------------------------
// SpinalCordCircuit on MI455X (gfx1250, wave32, WMMA).
//   K0  : weight repack/convert f32 -> f16 (conv filters im2col-major, linear
//         weights row-major [out][in])
//   K1  : causal conv1 (12->128, K=5) as 16x16x32 f16 WMMA implicit GEMM
//   K2  : causal conv2 (128->256, K=5) + the four lin_seq projections fused,
//         all WMMA; emits 8 time-major f16 input streams for the scan
//   K3  : persistent single-workgroup recurrent scan, 2000 steps, all
//         matmuls via v_wmma_f32_16x16x32_f16, states + buffers in 304KB LDS,
//         per-step inputs fetched with TDM tensor_load_to_lds (TENSORcnt).
// Workspace need ~165 MB (kin f16 31MB + SEQ f16 125MB + weights).
// ---------------------------------------------------------------------------

#define TLEN  2000
#define BATCH 64

typedef __attribute__((ext_vector_type(16))) _Float16 v16h;
typedef __attribute__((ext_vector_type(8)))  _Float16 v8h;
typedef __attribute__((ext_vector_type(8)))  float    v8f;
typedef __attribute__((ext_vector_type(4)))  float    v4f;
typedef __attribute__((ext_vector_type(4)))  unsigned u32x4;
typedef __attribute__((ext_vector_type(8)))  int      i32x8;
typedef __attribute__((ext_vector_type(4)))  int      i32x4;

__device__ __forceinline__ v8f wmma16(v16h a, v16h b, v8f c) {
  return __builtin_amdgcn_wmma_f32_16x16x32_f16(false, a, false, b, (short)0, c, false, false);
}

__device__ __forceinline__ float sigmoidf_(float x) { return 1.0f / (1.0f + __expf(-x)); }

// A fragment (16x32 f16, row-major source). lane: m = lane&15, kh = lane>>4.
// layout: a[i] = A[m][k0 + kh*16 + i]  (i=0..15)
__device__ __forceinline__ v16h a_frag_f16(const _Float16* A, int lda, int m, int kh, int k0) {
  const _Float16* r = A + (size_t)m * lda + k0 + kh * 16;
  v8h lo = *(const v8h*)(r);
  v8h hi = *(const v8h*)(r + 8);
  v16h a;
#pragma unroll
  for (int i = 0; i < 8; i++) { a[i] = lo[i]; a[8 + i] = hi[i]; }
  return a;
}

__device__ __forceinline__ v16h a_frag_f32(const float* A, int lda, int m, int kh, int k0) {
  const float* r = A + (size_t)m * lda + k0 + kh * 16;
  v4f x0 = *(const v4f*)(r);
  v4f x1 = *(const v4f*)(r + 4);
  v4f y0 = *(const v4f*)(r + 8);
  v4f y1 = *(const v4f*)(r + 12);
  v16h a;
#pragma unroll
  for (int i = 0; i < 4; i++) {
    a[i]      = (_Float16)x0[i];
    a[4 + i]  = (_Float16)x1[i];
    a[8 + i]  = (_Float16)y0[i];
    a[12 + i] = (_Float16)y1[i];
  }
  return a;
}

// B fragment (32x16 f16) from weights stored [N][K] row-major:
// lane: n = lane&15, 16 contiguous f16 at K = k0 + (lane>>4)*16.
__device__ __forceinline__ v16h b_frag(const _Float16* W, int ldb, int n, int kh, int k0) {
  return *(const v16h*)(W + (size_t)n * ldb + k0 + kh * 16);
}

// C[64 x 16*Ntiles] (f32, LDS) = A[64x64](f32 LDS) x B^T + bias.
// Tiles distributed over 16 waves.
__device__ __forceinline__ void gemm64(const float* A, const _Float16* B, int ldb,
                                       const float* bias, float* C, int ldc,
                                       int Ntiles, int wave, int lane) {
  const int nl = lane & 15, kh = lane >> 4;
  for (int tile = wave; tile < 4 * Ntiles; tile += 16) {
    const int mt = tile / Ntiles, nt = tile - mt * Ntiles;
    const int m0 = mt * 16, n0 = nt * 16;
    v16h a0 = a_frag_f32(A, 64, m0 + nl, kh, 0);
    v16h a1 = a_frag_f32(A, 64, m0 + nl, kh, 32);
    v16h b0 = b_frag(B, ldb, n0 + nl, kh, 0);
    v16h b1 = b_frag(B, ldb, n0 + nl, kh, 32);
    v8f acc = {};
    acc = wmma16(a0, b0, acc);
    acc = wmma16(a1, b1, acc);
    const float bb = bias ? bias[n0 + nl] : 0.0f;
#pragma unroll
    for (int v = 0; v < 8; v++)
      C[(size_t)(m0 + v + 8 * kh) * ldc + n0 + nl] = acc[v] + bb;
  }
}

// ---------------------------------------------------------------------------
// K0: weight prep
// ---------------------------------------------------------------------------
__global__ void sc_cvt_f16(const float* __restrict__ s, _Float16* __restrict__ d, int n) {
  int i = blockIdx.x * 256 + threadIdx.x;
  if (i < n) d[i] = (_Float16)s[i];
}

// w_k2c (128,12,5) -> W1T[128][64] f16, kk = ci*5+kt, pad 60..63 with 0
__global__ void sc_prep_w1(const float* __restrict__ w, _Float16* __restrict__ d) {
  int i = blockIdx.x * 256 + threadIdx.x;
  if (i >= 128 * 64) return;
  int ce = i >> 6, kk = i & 63;
  float v = 0.0f;
  if (kk < 60) { int ci = kk / 5, kt = kk - ci * 5; v = w[(ce * 12 + ci) * 5 + kt]; }
  d[i] = (_Float16)v;
}

// w_c2a (256,128,5) -> W2T[5][256][128] f16
__global__ void sc_prep_w2(const float* __restrict__ w, _Float16* __restrict__ d) {
  int i = blockIdx.x * 256 + threadIdx.x;
  if (i >= 5 * 256 * 128) return;
  int k = i / (256 * 128);
  int r = i - k * 256 * 128;
  int co = r >> 7, ci = r & 127;
  d[i] = (_Float16)w[(co * 128 + ci) * 5 + k];
}

// ---------------------------------------------------------------------------
// K1: conv1 + relu -> kin[b][t][128] f16  (implicit GEMM, K=60 padded to 64)
// ---------------------------------------------------------------------------
struct Conv1Args { const float* x; const _Float16* W1T; const float* b1; _Float16* kin; };

__global__ __launch_bounds__(256) void sc_conv1(Conv1Args A) {
  __shared__ _Float16 AT[16 * 64];  // im2col tile [t'][kk]
  const int blk = blockIdx.x;
  const int b = blk / (TLEN / 16);
  const int t0 = (blk - b * (TLEN / 16)) * 16;
  const int tid = threadIdx.x;
  for (int e = tid; e < 16 * 64; e += 256) {
    int m = e >> 6, kk = e & 63;
    float v = 0.0f;
    if (kk < 60) {
      int ci = kk / 5, kt = kk - ci * 5;
      int tt = t0 + m - 4 + kt;
      if (tt >= 0) v = A.x[((size_t)b * 12 + ci) * TLEN + tt];
    }
    AT[e] = (_Float16)v;
  }
  __syncthreads();
  const int wave = tid >> 5, lane = tid & 31;
  const int nl = lane & 15, kh = lane >> 4;
  const int n0 = wave * 16;  // 8 waves x 16 = 128 out channels
  v16h a0 = a_frag_f16(AT, 64, nl, kh, 0);
  v16h a1 = a_frag_f16(AT, 64, nl, kh, 32);
  v16h b0 = b_frag(A.W1T, 64, n0 + nl, kh, 0);
  v16h b1 = b_frag(A.W1T, 64, n0 + nl, kh, 32);
  v8f acc = {};
  acc = wmma16(a0, b0, acc);
  acc = wmma16(a1, b1, acc);
  const float bb = A.b1[n0 + nl];
#pragma unroll
  for (int v = 0; v < 8; v++) {
    int t = t0 + v + 8 * kh;
    float o = acc[v] + bb;
    o = o > 0.0f ? o : 0.0f;
    A.kin[((size_t)b * TLEN + t) * 128 + n0 + nl] = (_Float16)o;
  }
}

// ---------------------------------------------------------------------------
// K2: conv2 + relu (fused in LDS) + 4 lin_seq -> 8 scan input streams
//     SEQ[t][s][b][64] f16, s order: exF,iaiF1,iaiF2,mnF1,exE,iaiE1,iaiE2,mnE1
// ---------------------------------------------------------------------------
struct Conv2Args {
  const _Float16* kin; const _Float16* W2T; const float* b2; _Float16* SEQ;
  const _Float16* lw[8]; const float* lb[8]; int choff[8];
};

__global__ __launch_bounds__(256) void sc_conv2lin(Conv2Args A) {
  __shared__ _Float16 AFF[16 * 256];  // relu(aff) tile [t'][co]
  const int blk = blockIdx.x;
  const int b = blk / (TLEN / 16);
  const int t0 = (blk - b * (TLEN / 16)) * 16;
  const int tid = threadIdx.x, wave = tid >> 5, lane = tid & 31;
  const int nl = lane & 15, kh = lane >> 4;

  // ---- phase 1: conv2, each wave produces two co-tiles (32 channels) ----
  v8f acc0 = {}, acc1 = {};
  for (int k = 0; k < 5; k++) {
    const int tt = t0 + nl - 4 + k;  // input time for row m = nl
    const bool ok = (tt >= 0);
    const _Float16* row = A.kin + ((size_t)b * TLEN + (ok ? tt : 0)) * 128;
#pragma unroll
    for (int cb = 0; cb < 4; cb++) {
      v16h a = {};
      if (ok) {
        v8h lo = *(const v8h*)(row + cb * 32 + kh * 16);
        v8h hi = *(const v8h*)(row + cb * 32 + 8 + kh * 16);
#pragma unroll
        for (int i = 0; i < 8; i++) { a[i] = lo[i]; a[8 + i] = hi[i]; }
      }
      {
        int co = wave * 32 + nl;
        v16h bf = *(const v16h*)(A.W2T + ((size_t)k * 256 + co) * 128 + cb * 32 + kh * 16);
        acc0 = wmma16(a, bf, acc0);
      }
      {
        int co = wave * 32 + 16 + nl;
        v16h bf = *(const v16h*)(A.W2T + ((size_t)k * 256 + co) * 128 + cb * 32 + kh * 16);
        acc1 = wmma16(a, bf, acc1);
      }
    }
  }
  {
    int co = wave * 32 + nl;
    float bb = A.b2[co];
#pragma unroll
    for (int v = 0; v < 8; v++) {
      float o = acc0[v] + bb; o = o > 0.0f ? o : 0.0f;
      AFF[(v + 8 * kh) * 256 + co] = (_Float16)o;
    }
    co = wave * 32 + 16 + nl;
    bb = A.b2[co];
#pragma unroll
    for (int v = 0; v < 8; v++) {
      float o = acc1[v] + bb; o = o > 0.0f ? o : 0.0f;
      AFF[(v + 8 * kh) * 256 + co] = (_Float16)o;
    }
  }
  __syncthreads();

  // ---- phase 2: 8 streams x 4 n-tiles = 32 tiles over 8 waves ----
  for (int it = 0; it < 4; ++it) {
    const int tile = wave + 8 * it;
    const int s = tile >> 2, ntl = tile & 3, n0 = ntl * 16;
    const _Float16* Abase = AFF + A.choff[s];
    v16h a0 = a_frag_f16(Abase, 256, nl, kh, 0);
    v16h a1 = a_frag_f16(Abase, 256, nl, kh, 32);
    const _Float16* W = A.lw[s];
    v16h b0 = b_frag(W, 64, n0 + nl, kh, 0);
    v16h b1 = b_frag(W, 64, n0 + nl, kh, 32);
    v8f acc = {};
    acc = wmma16(a0, b0, acc);
    acc = wmma16(a1, b1, acc);
    const float bb = A.lb[s][n0 + nl];
#pragma unroll
    for (int v = 0; v < 8; v++) {
      int t = t0 + v + 8 * kh;
      A.SEQ[(((size_t)t * 8 + s) * 64 + b) * 64 + n0 + nl] = (_Float16)(acc[v] + bb);
    }
  }
}

// ---------------------------------------------------------------------------
// K3: persistent recurrent scan (1 workgroup, 512 threads = 16 waves)
// LDS map (bytes):
//   H[6][64][64] f32      @ 0       (98304)
//   CROSS_F[64][128] f32  @ 98304   (32768)
//   CROSS_E[64][128] f32  @ 131072  (32768)
//   HH[64][128] f32       @ 163840  (32768)
//   GX[64][64] f32        @ 196608  (16384)
//   XACT[64][64] f32      @ 212992  (16384)
//   LIN[64][64] f32       @ 229376  (16384)
//   SEQBUF[8][64][64] f16 @ 245760  (65536)   total 311296
// ---------------------------------------------------------------------------
struct LayerCfg {
  const float *gamma, *beta, *lnh_w, *lnh_b, *lnx_w, *lnx_b;
  const _Float16 *Wh, *Wx;
  int hidx, P;
  int kind[3];  // 0=SEQ stream, 1=CROSS_F+off, 2=CROSS_E+off, 3=LIN
  int off[3];
};
struct ScanArgs {
  LayerCfg L[6];
  const _Float16* WcrossF; const float* bcrossF;   // w_IaiE2 / b_IaiE2
  const _Float16* WcrossE; const float* bcrossE;   // w_IaiF2 / b_IaiF2
  const _Float16* WexF; const float* bexF;
  const _Float16* WexE; const float* bexE;
  const _Float16* SEQ;
  float* out;
};

__global__ __launch_bounds__(512) void sc_scan(ScanArgs A) {
  extern __shared__ char smem[];
  float*    H   = (float*)(smem);
  float*    CF  = (float*)(smem + 98304);
  float*    CE  = (float*)(smem + 131072);
  float*    HH  = (float*)(smem + 163840);
  float*    GX  = (float*)(smem + 196608);
  float*    XA  = (float*)(smem + 212992);
  float*    LIN = (float*)(smem + 229376);
  _Float16* SB  = (_Float16*)(smem + 245760);

  const int tid = threadIdx.x, wave = tid >> 5, lane = tid & 31;
  for (int i = tid; i < 6 * 4096; i += 512) H[i] = 0.0f;
  __syncthreads();

  const unsigned sb_lds = (unsigned)(size_t)(void*)SB;  // LDS byte offset (low 32 of flat)

#pragma unroll 1
  for (int t = 0; t < TLEN; ++t) {
    // ---- fetch SEQ[t] (64KB) into LDS ----
    const _Float16* src = A.SEQ + (size_t)t * (8 * 64 * 64);
#if __has_builtin(__builtin_amdgcn_tensor_load_to_lds)
    if (wave == 0) {
      // 1-row TDM descriptor: 8192 elements of 8 bytes.
      unsigned long long ga = (unsigned long long)(size_t)src;
      u32x4 g0;
      g0[0] = 1u;                                            // count=1, user mode
      g0[1] = sb_lds;                                        // lds_addr
      g0[2] = (unsigned)(ga & 0xFFFFFFFFu);                  // global_addr[31:0]
      g0[3] = (unsigned)((ga >> 32) & 0x1FFFFFFu) | (2u << 30);  // [56:32], type=2
      i32x8 g1;
      g1[0] = (3 << 16);                // wg_mask=0, data_size=8B
      g1[1] = (int)(0x2000u << 16);     // tensor_dim0 = 8192 (low16 in [31:16])
      g1[2] = (int)(1u << 16);          // tensor_dim0 hi=0, tensor_dim1 = 1
      g1[3] = (int)(0x2000u << 16);     // tensor_dim1 hi=0, tile_dim0 = 8192
      g1[4] = 1;                        // tile_dim1 = 1, tile_dim2 = 0
      g1[5] = 8192;                     // tensor_dim0_stride (low 32)
      g1[6] = 0;
      g1[7] = 0;
      i32x4 z4 = {0, 0, 0, 0};
      i32x8 z8 = {0, 0, 0, 0, 0, 0, 0, 0};
      __builtin_amdgcn_tensor_load_to_lds(g0, g1, z4, z4, z8, 0);
      __builtin_amdgcn_s_wait_tensorcnt(0);
    }
#else
    for (int i = tid; i < 8 * 64 * 64 / 8; i += 512)
      ((v8h*)SB)[i] = ((const v8h*)src)[i];
#endif
    __syncthreads();

    // ---- cross projections from previous-step states ----
    gemm64(H + 4 * 4096, A.WcrossF, 64, A.bcrossF, CF, 128, 8, wave, lane);  // h_IaiE
    gemm64(H + 1 * 4096, A.WcrossE, 64, A.bcrossE, CE, 128, 8, wave, lane);  // h_IaiF
    __syncthreads();

#pragma unroll 1
    for (int li = 0; li < 6; ++li) {
      const LayerCfg& L = A.L[li];
      // hh = hx @ Wh.T (no bias)
      gemm64(H + (size_t)L.hidx * 4096, L.Wh, 64, nullptr, HH, 128, 8, wave, lane);
      if (li == 2) gemm64(H + 0 * 4096, A.WexF, 64, A.bexF, LIN, 64, 4, wave, lane);  // lin(n_exF)
      if (li == 5) gemm64(H + 3 * 4096, A.WexE, 64, A.bexE, LIN, 64, 4, wave, lane);  // lin(n_exE)
      __syncthreads();

      // LayerNorm over 128 features of HH, in place
      {
        const int b = tid >> 3, j = tid & 7;
        float vals[16], s = 0.0f, ss = 0.0f;
#pragma unroll
        for (int i = 0; i < 16; i++) {
          float v = HH[b * 128 + j * 16 + i];
          vals[i] = v; s += v; ss += v * v;
        }
        for (int m = 1; m < 8; m <<= 1) { s += __shfl_xor(s, m, 32); ss += __shfl_xor(ss, m, 32); }
        const float mean = s * (1.0f / 128.0f);
        const float var = ss * (1.0f / 128.0f) - mean * mean;
        const float r = rsqrtf(var + 1e-5f);
#pragma unroll
        for (int i = 0; i < 16; i++) {
          int f = j * 16 + i;
          HH[b * 128 + f] = (vals[i] - mean) * r * L.lnh_w[f] + L.lnh_b[f];
        }
      }
      __syncthreads();

      // in_layer + xact
      {
        const int b = tid >> 3, j = tid & 7;
#pragma unroll
        for (int i = 0; i < 8; i++) {
          int f = j * 8 + i;
          float h = H[(size_t)L.hidx * 4096 + b * 64 + f];
          float acc = 0.0f;
          for (int p = 0; p < L.P; ++p) {
            float g = sigmoidf_(L.gamma[p * 64 + f] * h + L.beta[p * 64 + f]);
            float x;
            int k = L.kind[p], o = L.off[p];
            if (k == 0)      x = (float)SB[((size_t)o * 64 + b) * 64 + f];
            else if (k == 1) x = CF[b * 128 + o + f];
            else if (k == 2) x = CE[b * 128 + o + f];
            else             x = LIN[b * 64 + f];
            acc += g * x;
          }
          float xa = acc + HH[b * 128 + f];
          XA[b * 64 + f] = xa > 0.0f ? xa : 0.0f;
        }
      }
      __syncthreads();

      // gx = xact @ Wx.T (no bias)
      gemm64(XA, L.Wx, 64, nullptr, GX, 64, 4, wave, lane);
      __syncthreads();

      // gg + state update (+ output for motoneuron layers)
      {
        const int b = tid >> 3, j = tid & 7;
        float vals[8], s = 0.0f, ss = 0.0f;
#pragma unroll
        for (int i = 0; i < 8; i++) {
          float v = GX[b * 64 + j * 8 + i];
          vals[i] = v; s += v; ss += v * v;
        }
        for (int m = 1; m < 8; m <<= 1) { s += __shfl_xor(s, m, 32); ss += __shfl_xor(ss, m, 32); }
        const float mean = s * (1.0f / 64.0f);
        const float var = ss * (1.0f / 64.0f) - mean * mean;
        const float r = rsqrtf(var + 1e-5f);
#pragma unroll
        for (int i = 0; i < 8; i++) {
          int f = j * 8 + i;
          float gn = (vals[i] - mean) * r * L.lnx_w[f] + L.lnx_b[f];
          float gg = sigmoidf_(gn + HH[b * 128 + 64 + f]);
          float h = H[(size_t)L.hidx * 4096 + b * 64 + f];
          float x = XA[b * 64 + f];
          float hn = (1.0f - gg) * h + gg * x;
          H[(size_t)L.hidx * 4096 + b * 64 + f] = hn;
          if (li == 2) A.out[((size_t)b * 128 + f) * TLEN + t] = hn;
          if (li == 5) A.out[((size_t)b * 128 + 64 + f) * TLEN + t] = hn;
        }
      }
      __syncthreads();
    }
  }
}

// ---------------------------------------------------------------------------
// Host launcher
// ---------------------------------------------------------------------------
extern "C" void kernel_launch(void* const* d_in, const int* in_sizes, int n_in,
                              void* d_out, int out_size, void* d_ws, size_t ws_size,
                              hipStream_t stream) {
  (void)in_sizes; (void)n_in; (void)out_size; (void)ws_size;
  auto F = [&](int i) -> const float* { return (const float*)d_in[i]; };
  char* ws = (char*)d_ws;
  size_t off = 0;
  auto alloc = [&](size_t bytes) -> size_t {
    size_t o = off; off += (bytes + 255) & ~(size_t)255; return o;
  };

  const size_t off_kin = alloc((size_t)BATCH * TLEN * 128 * 2);
  const size_t off_seq = alloc((size_t)TLEN * 8 * 64 * 64 * 2);
  const size_t off_w1t = alloc(128 * 64 * 2);
  const size_t off_w2t = alloc((size_t)5 * 256 * 128 * 2);
  size_t off_wlin[4];
  for (int i = 0; i < 4; i++) off_wlin[i] = alloc(128 * 64 * 2);  // WF1,WF2,WE1,WE2
  const size_t off_wcF = alloc(128 * 64 * 2);   // w_IaiE2 f16
  const size_t off_wcE = alloc(128 * 64 * 2);   // w_IaiF2 f16
  const size_t off_wexF = alloc(64 * 64 * 2);
  const size_t off_wexE = alloc(64 * 64 * 2);
  size_t off_wh[6], off_wx[6];
  for (int l = 0; l < 6; l++) { off_wh[l] = alloc(128 * 64 * 2); off_wx[l] = alloc(64 * 64 * 2); }

  // ---- weight prep ----
  struct CvtJob { int idx; size_t dst; int n; };
  const CvtJob jobs[] = {
    {5,  off_wlin[0], 128 * 64}, {7,  off_wlin[1], 128 * 64},
    {9,  off_wlin[2], 128 * 64}, {11, off_wlin[3], 128 * 64},
    {19, off_wcF, 128 * 64},     {17, off_wcE, 128 * 64},
    {13, off_wexF, 64 * 64},     {15, off_wexE, 64 * 64},
  };
  for (const auto& j : jobs)
    sc_cvt_f16<<<(j.n + 255) / 256, 256, 0, stream>>>(F(j.idx), (_Float16*)(ws + j.dst), j.n);
  for (int l = 0; l < 6; l++) {
    const int base = 21 + 8 * l;
    sc_cvt_f16<<<(128 * 64 + 255) / 256, 256, 0, stream>>>(F(base + 5), (_Float16*)(ws + off_wh[l]), 128 * 64);
    sc_cvt_f16<<<(64 * 64 + 255) / 256, 256, 0, stream>>>(F(base + 2), (_Float16*)(ws + off_wx[l]), 64 * 64);
  }
  sc_prep_w1<<<(128 * 64 + 255) / 256, 256, 0, stream>>>(F(1), (_Float16*)(ws + off_w1t));
  sc_prep_w2<<<(5 * 256 * 128 + 255) / 256, 256, 0, stream>>>(F(3), (_Float16*)(ws + off_w2t));

  // ---- conv1 ----
  Conv1Args c1;
  c1.x = F(0);
  c1.W1T = (const _Float16*)(ws + off_w1t);
  c1.b1 = F(2);
  c1.kin = (_Float16*)(ws + off_kin);
  sc_conv1<<<BATCH * (TLEN / 16), 256, 0, stream>>>(c1);

  // ---- conv2 + lin_seq ----
  Conv2Args c2;
  c2.kin = (const _Float16*)(ws + off_kin);
  c2.W2T = (const _Float16*)(ws + off_w2t);
  c2.b2 = F(4);
  c2.SEQ = (_Float16*)(ws + off_seq);
  // stream s -> which lin weight (0=WF1/IaF, 1=WF2/IIF, 2=WE1/IaE, 3=WE2/IIE), row offset
  const int widx[8]  = {1, 0, 1, 0, 3, 2, 3, 2};
  const int roff[8]  = {64, 0, 0, 64, 64, 0, 0, 64};
  const int bidx[4]  = {6, 8, 10, 12};            // b_IaF2, b_IIF2, b_IaE2, b_IIE2
  const int choff[4] = {0, 64, 128, 192};         // aff channel block per weight
  for (int s = 0; s < 8; s++) {
    const int w = widx[s];
    c2.lw[s] = (const _Float16*)(ws + off_wlin[w]) + roff[s] * 64;
    c2.lb[s] = F(bidx[w]) + roff[s];
    c2.choff[s] = choff[w];
  }
  sc_conv2lin<<<BATCH * (TLEN / 16), 256, 0, stream>>>(c2);

  // ---- scan ----
  ScanArgs sa;
  const int kinds[6][3] = {{0,0,0},{0,0,1},{0,3,1},{0,0,0},{0,0,2},{0,3,2}};
  const int offs [6][3] = {{0,0,0},{1,2,0},{3,0,64},{4,0,0},{5,6,0},{7,0,64}};
  const int Ps[6] = {1, 3, 3, 1, 3, 3};
  for (int l = 0; l < 6; l++) {
    const int base = 21 + 8 * l;
    LayerCfg& L = sa.L[l];
    L.gamma = F(base + 0); L.beta = F(base + 1);
    L.lnx_w = F(base + 3); L.lnx_b = F(base + 4);
    L.lnh_w = F(base + 6); L.lnh_b = F(base + 7);
    L.Wh = (const _Float16*)(ws + off_wh[l]);
    L.Wx = (const _Float16*)(ws + off_wx[l]);
    L.hidx = l; L.P = Ps[l];
    for (int p = 0; p < 3; p++) { L.kind[p] = kinds[l][p]; L.off[p] = offs[l][p]; }
  }
  sa.WcrossF = (const _Float16*)(ws + off_wcF); sa.bcrossF = F(20);  // w_IaiE2/b_IaiE2
  sa.WcrossE = (const _Float16*)(ws + off_wcE); sa.bcrossE = F(18);  // w_IaiF2/b_IaiF2
  sa.WexF = (const _Float16*)(ws + off_wexF); sa.bexF = F(14);
  sa.WexE = (const _Float16*)(ws + off_wexE); sa.bexE = F(16);
  sa.SEQ = (const _Float16*)(ws + off_seq);
  sa.out = (float*)d_out;
  sc_scan<<<1, 512, 311296, stream>>>(sa);
}
